// CoAtNetTransformerBlock_54477365182957
// MI455X (gfx1250) — matrix operand order repaired and checked by reading, not compiled, not run on hardware
//
#include <hip/hip_runtime.h>
#include <hip/hip_fp16.h>
#include <math.h>

typedef __attribute__((ext_vector_type(16))) _Float16 v16h;
typedef __attribute__((ext_vector_type(8)))  _Float16 v8h;
typedef __attribute__((ext_vector_type(8)))  float    v8f;
typedef int async_int4 __attribute__((vector_size(16)));   // matches builtin param type

#define WMMA_F16(a,b,c) __builtin_amdgcn_wmma_f32_16x16x32_f16(false,(a),false,(b),(short)0,(c),false,false)

// ---- CDNA5 async global->LDS path (guarded: falls back to sync copies) ----
#if defined(__has_builtin)
#if __has_builtin(__builtin_amdgcn_global_load_async_to_lds_b128) && \
    __has_builtin(__builtin_amdgcn_s_wait_asynccnt)
#define HAVE_ASYNC_LDS 1
#endif
#endif
#ifndef HAVE_ASYNC_LDS
#define HAVE_ASYNC_LDS 0
#endif

#if HAVE_ASYNC_LDS
// signature (from hipcc diagnostic): (global int4* src, lds int4* dst, imm offset, imm cpol)
#define ASYNC_CP16(dstLds, srcGlb)                                              \
    __builtin_amdgcn_global_load_async_to_lds_b128(                             \
        (__attribute__((address_space(1))) async_int4*)(srcGlb),                \
        (__attribute__((address_space(3))) async_int4*)(dstLds), 0, 0)
#endif

// ---------------------------------------------------------------------------
// Fragment loaders (per CDNA5 ISA 7.12.2 VGPR layouts, wave32)
// A (16x32 f16): lane(l,g) -> row M=l, halves j: K = j + 8g (+8 if j>=8)
// B (32x16 f16): lane(l,g) -> col N=l, halves j: K = j + 8g (+8 if j>=8)
// Stored in LDS row-major [row][K], ld multiple of 8 halves (16B align).
// ---------------------------------------------------------------------------
static __device__ __forceinline__ v16h frag_a(const _Float16* base, int ld, int row0, int k0, int l, int g) {
    const _Float16* p = base + (size_t)(row0 + l) * ld + k0 + 8 * g;
    v8h lo = *(const v8h*)p;
    v8h hi = *(const v8h*)(p + 16);
    return __builtin_shufflevector(lo, hi, 0,1,2,3,4,5,6,7,8,9,10,11,12,13,14,15);
}
static __device__ __forceinline__ v16h frag_b(const _Float16* base, int ld, int row0, int k0, int l, int g) {
    const _Float16* p = base + (size_t)(row0 + l) * ld + k0 + 8 * g;
    v8h lo = *(const v8h*)p;
    v8h hi = *(const v8h*)(p + 16);
    return __builtin_shufflevector(lo, hi, 0,1,2,3,4,5,6,7,8,9,10,11,12,13,14,15);
}

// ---------------------------------------------------------------------------
// Transpose + convert: src f32 [K][N] -> dst f16 [N][K].  K,N multiples of 32.
// ---------------------------------------------------------------------------
__global__ __launch_bounds__(256) void tconv_kernel(const float* __restrict__ src,
                                                    _Float16* __restrict__ dst,
                                                    int K, int N) {
    __shared__ float tile[32][33];
    const int n0 = blockIdx.x * 32, k0 = blockIdx.y * 32;
    const int tx = threadIdx.x & 31, ty = threadIdx.x >> 5;
    #pragma unroll
    for (int i = 0; i < 4; i++)
        tile[ty + 8 * i][tx] = src[(size_t)(k0 + ty + 8 * i) * N + n0 + tx];
    __syncthreads();
    #pragma unroll
    for (int i = 0; i < 4; i++)
        dst[(size_t)(n0 + ty + 8 * i) * K + k0 + tx] = (_Float16)tile[tx][ty + 8 * i];
}

__global__ void packb_kernel(const float* __restrict__ bq, const float* __restrict__ bk,
                             const float* __restrict__ bv, float* __restrict__ dst) {
    int i = blockIdx.x * 256 + threadIdx.x;
    if (i < 768) { dst[i] = bq[i]; dst[768 + i] = bk[i]; dst[1536 + i] = bv[i]; }
}

// ---------------------------------------------------------------------------
// Fused NCHW->token-major transpose + LayerNorm.
// grid (13 n-tiles, 64 batches), 256 threads. Dynamic LDS: 768*17 floats.
// ---------------------------------------------------------------------------
__global__ __launch_bounds__(256) void ln_kernel(const float* __restrict__ x,
                                                 const float* __restrict__ gam,
                                                 const float* __restrict__ bet,
                                                 _Float16* __restrict__ xn16,
                                                 float* __restrict__ xtok) {
    extern __shared__ float tile[];               // [768][17]
    __shared__ float p1[16][17], p2[16][17], smu[16], srs[16];
    const int b = blockIdx.y, n0 = blockIdx.x * 16;
    const int t = threadIdx.x;
    {   // load 16 tokens x 768 channels, coalesced along n
        const int j = t & 15, cs = t >> 4;
        const int n = n0 + j;
        for (int i = 0; i < 48; i++) {
            const int c = i * 16 + cs;
            float v = 0.0f;
            if (n < 196) v = x[((size_t)b * 768 + c) * 196 + n];
            tile[c * 17 + j] = v;
        }
    }
    __syncthreads();
    {   // partial sums: 16 partials per token
        const int s = t >> 4, j = t & 15;
        float a = 0.f, q = 0.f;
        for (int i = 0; i < 48; i++) { float v = tile[(s * 48 + i) * 17 + j]; a += v; q += v * v; }
        p1[s][j] = a; p2[s][j] = q;
    }
    __syncthreads();
    if (t < 16) {
        float a = 0.f, q = 0.f;
        for (int s = 0; s < 16; s++) { a += p1[s][t]; q += p2[s][t]; }
        float mu = a * (1.0f / 768.0f);
        float var = q * (1.0f / 768.0f) - mu * mu;
        smu[t] = mu; srs[t] = rsqrtf(var + 1e-5f);
    }
    __syncthreads();
    {   // normalize + write token-major (coalesced along c)
        const int j = t >> 4, cl = t & 15;
        const int n = n0 + j;
        if (n < 196) {
            const size_t tok = (size_t)b * 196 + n;
            const float mu = smu[j], rs = srs[j];
            for (int i = 0; i < 48; i++) {
                const int c = i * 16 + cl;
                float v = (tile[c * 17 + j] - mu) * rs * gam[c] + bet[c];
                xn16[tok * 768 + c] = (_Float16)v;
                xtok[tok * 768 + c] = v;
            }
        }
    }
}

// ---------------------------------------------------------------------------
// Generic WMMA GEMM: C[M,N] = A[M,K] * B^T[N,K].
// Block 128x128, BK=64, double-buffered LDS (dyn, 4*128*72*2 = 73728 B),
// async global->LDS prefetch of tile i+1 overlapped with WMMAs on tile i.
// 8 waves, each wave 32x64 (2x4 tiles, 16 WMMAs per K-step).
// MODE 0: +bias -> f16            (QKV)
// MODE 1: +bias+resid -> f32+f16  (WO + residual)
// MODE 2: +bias, GELU -> f16      (MLP fc1)
// MODE 3: +bias+resid -> f32 NCHW (MLP fc2 + residual, direct to d_out)
// ---------------------------------------------------------------------------
template <int MODE>
__global__ __launch_bounds__(256) void gemm_wmma(const _Float16* __restrict__ A,
                                                 const _Float16* __restrict__ BT,
                                                 const float* __restrict__ bias,
                                                 const float* __restrict__ resid,
                                                 _Float16* __restrict__ out16,
                                                 float* __restrict__ out32,
                                                 int M, int N, int K) {
    extern __shared__ _Float16 smem16[];
    const int ABUF = 128 * 72;          // halves per matrix buffer
    const int BUF  = 2 * ABUF;          // halves per (A+B) double-buffer step

    const int m0 = blockIdx.x * 128, n0 = blockIdx.y * 128;
    const int t = threadIdx.x, lane = t & 31;
    const int l = lane & 15, g = lane >> 4, wv = t >> 5;
    const int wm = (wv & 3) * 32, wn = (wv >> 2) * 64;
    v8f acc[2][4] = {};

    const int lrow = t >> 1;                 // 0..127
    const int lkh = (t & 1) * 32;            // halves within 64-wide K slab
    const _Float16* gA = A + (size_t)(m0 + lrow) * K + lkh;
    const _Float16* gB = BT + (size_t)(n0 + lrow) * K + lkh;
    const int soff = lrow * 72 + lkh;
    const int nIter = K >> 6;

    auto compute = [&](const _Float16* cA, const _Float16* cB) {
        v16h a00 = frag_a(cA, 72, wm,       0, l, g);
        v16h a01 = frag_a(cA, 72, wm,      32, l, g);
        v16h a10 = frag_a(cA, 72, wm + 16,  0, l, g);
        v16h a11 = frag_a(cA, 72, wm + 16, 32, l, g);
        #pragma unroll
        for (int j = 0; j < 4; j++) {
            v16h b0f = frag_b(cB, 72, wn + j * 16,  0, l, g);
            v16h b1f = frag_b(cB, 72, wn + j * 16, 32, l, g);
            acc[0][j] = WMMA_F16(a00, b0f, acc[0][j]);
            acc[0][j] = WMMA_F16(a01, b1f, acc[0][j]);
            acc[1][j] = WMMA_F16(a10, b0f, acc[1][j]);
            acc[1][j] = WMMA_F16(a11, b1f, acc[1][j]);
        }
    };

#if HAVE_ASYNC_LDS
    #pragma unroll
    for (int u = 0; u < 4; u++) {
        ASYNC_CP16(smem16 + soff + u * 8,        gA + u * 8);
        ASYNC_CP16(smem16 + ABUF + soff + u * 8, gB + u * 8);
    }
    for (int i = 0; i < nIter; i++) {
        const int co = (i & 1) * BUF;            // current buffer offset
        if (i + 1 < nIter) {
            const int po = ((i + 1) & 1) * BUF;  // prefetch buffer offset
            const int kk = (i + 1) << 6;
            #pragma unroll
            for (int u = 0; u < 4; u++) {
                ASYNC_CP16(smem16 + po + soff + u * 8,        gA + kk + u * 8);
                ASYNC_CP16(smem16 + po + ABUF + soff + u * 8, gB + kk + u * 8);
            }
            __builtin_amdgcn_s_wait_asynccnt(8);   // tile i resident; tile i+1 in flight
        } else {
            __builtin_amdgcn_s_wait_asynccnt(0);
        }
        __syncthreads();
        compute(smem16 + co, smem16 + co + ABUF);
        __syncthreads();
    }
#else
    uint4 ra[4], rb[4];
    #pragma unroll
    for (int u = 0; u < 4; u++) {
        ra[u] = *(const uint4*)(gA + u * 8);
        rb[u] = *(const uint4*)(gB + u * 8);
    }
    #pragma unroll
    for (int u = 0; u < 4; u++) {
        *(uint4*)(smem16 + soff + u * 8)        = ra[u];
        *(uint4*)(smem16 + ABUF + soff + u * 8) = rb[u];
    }
    __syncthreads();
    for (int i = 0; i < nIter; i++) {
        const int co = (i & 1) * BUF;
        if (i + 1 < nIter) {
            const int kk = (i + 1) << 6;
            #pragma unroll
            for (int u = 0; u < 4; u++) {
                ra[u] = *(const uint4*)(gA + kk + u * 8);
                rb[u] = *(const uint4*)(gB + kk + u * 8);
            }
        }
        compute(smem16 + co, smem16 + co + ABUF);
        if (i + 1 < nIter) {
            const int po = ((i + 1) & 1) * BUF;
            __syncthreads();
            #pragma unroll
            for (int u = 0; u < 4; u++) {
                *(uint4*)(smem16 + po + soff + u * 8)        = ra[u];
                *(uint4*)(smem16 + po + ABUF + soff + u * 8) = rb[u];
            }
            __syncthreads();
        }
    }
#endif

    #pragma unroll
    for (int i = 0; i < 2; i++) {
        #pragma unroll
        for (int j = 0; j < 4; j++) {
            const int n = n0 + wn + j * 16 + l;
            const float bn = bias[n];
            #pragma unroll
            for (int r = 0; r < 8; r++) {
                const int m = m0 + wm + i * 16 + 8 * g + r;
                float v = acc[i][j][r] + bn;
                if (MODE == 1 || MODE == 3) v += resid[(size_t)m * N + n];
                if (MODE == 2) v = 0.5f * v * (1.0f + erff(v * 0.70710678118654752f));
                if (MODE == 0 || MODE == 2) out16[(size_t)m * N + n] = (_Float16)v;
                if (MODE == 1) {
                    out32[(size_t)m * N + n] = v;
                    out16[(size_t)m * N + n] = (_Float16)v;
                }
                if (MODE == 3) {
                    const int bb = m / 196, nn = m - bb * 196;
                    out32[((size_t)bb * 768 + n) * 196 + nn] = v;
                }
            }
        }
    }
}

// ---------------------------------------------------------------------------
// Attention: grid (12 heads, 64 batches), 416 threads = 13 waves.
// Wave w handles query rows [16w, 16w+16). N=196 padded to 208 (S) / 224 (PV).
// Dynamic LDS layout: Q[208][72], K[208][72], VT[64][232], P[13][16][232], bias[729].
// ---------------------------------------------------------------------------
__global__ __launch_bounds__(416) void attn_kernel(const _Float16* __restrict__ qkv,
                                                   const float* __restrict__ relb,
                                                   _Float16* __restrict__ out) {
    extern __shared__ char smem[];
    _Float16* sQ  = (_Float16*)smem;
    _Float16* sK  = sQ + 208 * 72;
    _Float16* sVT = sK + 208 * 72;
    _Float16* sP  = sVT + 64 * 232;
    float* sBias  = (float*)(sP + 13 * 16 * 232);
    const int h = blockIdx.x, b = blockIdx.y;
    const int t = threadIdx.x, lane = t & 31, wv = t >> 5;
    const int l = lane & 15, g = lane >> 4;

    for (int i = t; i < 64 * 232; i += 416) sVT[i] = (_Float16)0;
    for (int i = t; i < 13 * 16 * 232; i += 416) sP[i] = (_Float16)0;
    __syncthreads();

    for (int i = t; i < 729; i += 416) sBias[i] = relb[h * 729 + i];
    {   // stage Q, K rows; V transposed. 2 threads per row, 32 halves each.
        const int row = t >> 1, d0 = (t & 1) * 32;
        _Float16* dq = sQ + row * 72 + d0;
        _Float16* dk = sK + row * 72 + d0;
        if (row < 196) {
            const _Float16* gq = qkv + ((size_t)b * 196 + row) * 2304 + h * 64 + d0;
            #pragma unroll
            for (int u = 0; u < 4; u++) *(uint4*)(dq + u * 8) = *(const uint4*)(gq + u * 8);
            #pragma unroll
            for (int u = 0; u < 4; u++) *(uint4*)(dk + u * 8) = *(const uint4*)(gq + 768 + u * 8);
            const _Float16* gv = gq + 1536;
            #pragma unroll
            for (int u = 0; u < 4; u++) {
                v8h vvv = *(const v8h*)(gv + u * 8);
                #pragma unroll
                for (int e = 0; e < 8; e++) sVT[(d0 + u * 8 + e) * 232 + row] = vvv[e];
            }
        } else {
            uint4 z = {0, 0, 0, 0};
            #pragma unroll
            for (int u = 0; u < 4; u++) { *(uint4*)(dq + u * 8) = z; *(uint4*)(dk + u * 8) = z; }
        }
    }
    __syncthreads();

    // ---- S = Q K^T (13 key tiles, d=64 in two k=32 steps) ----
    v8f acc[13] = {};
    const int mrow = wv * 16;
    v16h q0 = frag_a(sQ, 72, mrow, 0, l, g);
    v16h q1 = frag_a(sQ, 72, mrow, 32, l, g);
    #pragma unroll
    for (int kt = 0; kt < 13; kt++) {
        v16h k0f = frag_b(sK, 72, kt * 16, 0, l, g);
        v16h k1f = frag_b(sK, 72, kt * 16, 32, l, g);
        acc[kt] = WMMA_F16(q0, k0f, acc[kt]);
        acc[kt] = WMMA_F16(q1, k1f, acc[kt]);
    }

    // ---- relative-position bias + mask ----
    #pragma unroll
    for (int r = 0; r < 8; r++) {
        const int m = mrow + 8 * g + r;
        const int qy = m / 14, qx = m - qy * 14;
        #pragma unroll
        for (int kt = 0; kt < 13; kt++) {
            const int n = kt * 16 + l;
            if (m < 196 && n < 196) {
                const int ky = n / 14, kx = n - ky * 14;
                acc[kt][r] += sBias[(ky - qy + 14) * 14 + (kx - qx + 14)];
            } else {
                acc[kt][r] = -1e30f;
            }
        }
    }

    // ---- softmax (row = 16 lanes of one half x 13 regs) + write P ----
    _Float16* pw = sP + wv * 16 * 232;
    #pragma unroll
    for (int r = 0; r < 8; r++) {
        float mx = -1e30f;
        #pragma unroll
        for (int kt = 0; kt < 13; kt++) mx = fmaxf(mx, acc[kt][r]);
        mx = fmaxf(mx, __shfl_xor(mx, 1, 32));
        mx = fmaxf(mx, __shfl_xor(mx, 2, 32));
        mx = fmaxf(mx, __shfl_xor(mx, 4, 32));
        mx = fmaxf(mx, __shfl_xor(mx, 8, 32));
        float s = 0.0f;
        #pragma unroll
        for (int kt = 0; kt < 13; kt++) { float e = __expf(acc[kt][r] - mx); acc[kt][r] = e; s += e; }
        s += __shfl_xor(s, 1, 32);
        s += __shfl_xor(s, 2, 32);
        s += __shfl_xor(s, 4, 32);
        s += __shfl_xor(s, 8, 32);
        const float inv = 1.0f / fmaxf(s, 1e-20f);
        const int prow = (r + 8 * g) * 232;
        #pragma unroll
        for (int kt = 0; kt < 13; kt++) pw[prow + kt * 16 + l] = (_Float16)(acc[kt][r] * inv);
    }
    __syncthreads();

    // ---- out = P V (K padded to 224 = 7x32) ----
    v8f o[4] = {};
    #pragma unroll
    for (int ks = 0; ks < 7; ks++) {
        v16h pf = frag_a(pw, 232, 0, ks * 32, l, g);
        #pragma unroll
        for (int j = 0; j < 4; j++) {
            v16h vf = frag_b(sVT, 232, j * 16, ks * 32, l, g);
            o[j] = WMMA_F16(pf, vf, o[j]);
        }
    }
    #pragma unroll
    for (int j = 0; j < 4; j++) {
        #pragma unroll
        for (int r = 0; r < 8; r++) {
            const int m = mrow + 8 * g + r;
            if (m < 196)
                out[((size_t)b * 196 + m) * 768 + h * 64 + j * 16 + l] = (_Float16)o[j][r];
        }
    }
}

// ---------------------------------------------------------------------------
extern "C" void kernel_launch(void* const* d_in, const int* in_sizes, int n_in,
                              void* d_out, int out_size, void* d_ws, size_t ws_size,
                              hipStream_t stream) {
    const float* x    = (const float*)d_in[0];
    const float* ln_g = (const float*)d_in[1];
    const float* ln_b = (const float*)d_in[2];
    const float* wq   = (const float*)d_in[3];
    const float* bq   = (const float*)d_in[4];
    const float* wk   = (const float*)d_in[5];
    const float* bk   = (const float*)d_in[6];
    const float* wvw  = (const float*)d_in[7];
    const float* bv   = (const float*)d_in[8];
    const float* wo   = (const float*)d_in[9];
    const float* bo   = (const float*)d_in[10];
    const float* w1   = (const float*)d_in[11];
    const float* b1   = (const float*)d_in[12];
    const float* w2   = (const float*)d_in[13];
    const float* b2   = (const float*)d_in[14];
    const float* relb = (const float*)d_in[15];

    char* ws = (char*)d_ws;
    size_t off = 0;
    auto alloc = [&](size_t bytes) -> char* {
        char* p = ws + off;
        off += (bytes + 255) & ~(size_t)255;
        return p;
    };
    _Float16* wqkvT = (_Float16*)alloc(2304ull * 768 * 2);
    _Float16* woT   = (_Float16*)alloc(768ull * 768 * 2);
    _Float16* w1T   = (_Float16*)alloc(3072ull * 768 * 2);
    _Float16* w2T   = (_Float16*)alloc(768ull * 3072 * 2);
    float*    bqkv  = (float*)alloc(2304 * 4);
    // region A: [xn16 | qkv16] during attention path, reused as h1 for MLP
    char*     regA  = alloc(12544ull * 3072 * 2);
    _Float16* xn16  = (_Float16*)regA;
    _Float16* qkv16 = (_Float16*)(regA + 12544ull * 768 * 2);
    _Float16* h1    = (_Float16*)regA;
    _Float16* attn16 = (_Float16*)alloc(12544ull * 768 * 2);
    float*    xtok   = (float*)alloc(12544ull * 768 * 4);
    float*    x2tok  = (float*)alloc(12544ull * 768 * 4);
    _Float16* x2_16  = (_Float16*)alloc(12544ull * 768 * 2);

    // 1) weight conversion (f32 [K][N] -> f16 [N][K])
    tconv_kernel<<<dim3(24, 24), 256, 0, stream>>>(wq,  wqkvT,                768, 768);
    tconv_kernel<<<dim3(24, 24), 256, 0, stream>>>(wk,  wqkvT + 768ull * 768, 768, 768);
    tconv_kernel<<<dim3(24, 24), 256, 0, stream>>>(wvw, wqkvT + 1536ull * 768, 768, 768);
    tconv_kernel<<<dim3(24, 24), 256, 0, stream>>>(wo,  woT, 768, 768);
    tconv_kernel<<<dim3(96, 24), 256, 0, stream>>>(w1,  w1T, 768, 3072);
    tconv_kernel<<<dim3(24, 96), 256, 0, stream>>>(w2,  w2T, 3072, 768);
    packb_kernel<<<3, 256, 0, stream>>>(bq, bk, bv, bqkv);

    // 2) LayerNorm + NCHW->token transpose
    size_t ln_lds = 768 * 17 * sizeof(float);
    (void)hipFuncSetAttribute((const void*)ln_kernel, hipFuncAttributeMaxDynamicSharedMemorySize, (int)ln_lds);
    ln_kernel<<<dim3(13, 64), 256, ln_lds, stream>>>(x, ln_g, ln_b, xn16, xtok);

    // GEMM dynamic LDS: 2 buffers x (A+B) x 128x72 halves
    size_t gemm_lds = (size_t)4 * 128 * 72 * 2;
    (void)hipFuncSetAttribute((const void*)gemm_wmma<0>, hipFuncAttributeMaxDynamicSharedMemorySize, (int)gemm_lds);
    (void)hipFuncSetAttribute((const void*)gemm_wmma<1>, hipFuncAttributeMaxDynamicSharedMemorySize, (int)gemm_lds);
    (void)hipFuncSetAttribute((const void*)gemm_wmma<2>, hipFuncAttributeMaxDynamicSharedMemorySize, (int)gemm_lds);
    (void)hipFuncSetAttribute((const void*)gemm_wmma<3>, hipFuncAttributeMaxDynamicSharedMemorySize, (int)gemm_lds);

    // 3) fused QKV projection
    gemm_wmma<0><<<dim3(98, 18), 256, gemm_lds, stream>>>(xn16, wqkvT, bqkv, nullptr, qkv16, nullptr,
                                                          12544, 2304, 768);

    // 4) attention
    size_t attn_lds = (size_t)(208 * 72 + 208 * 72 + 64 * 232 + 13 * 16 * 232) * 2 + 729 * 4;
    (void)hipFuncSetAttribute((const void*)attn_kernel, hipFuncAttributeMaxDynamicSharedMemorySize, (int)attn_lds);
    attn_kernel<<<dim3(12, 64), 416, attn_lds, stream>>>(qkv16, relb, attn16);

    // 5) output projection + residual
    gemm_wmma<1><<<dim3(98, 6), 256, gemm_lds, stream>>>(attn16, woT, bo, xtok, x2_16, x2tok,
                                                         12544, 768, 768);
    // 6) MLP fc1 + GELU
    gemm_wmma<2><<<dim3(98, 24), 256, gemm_lds, stream>>>(x2_16, w1T, b1, nullptr, h1, nullptr,
                                                          12544, 3072, 768);
    // 7) MLP fc2 + residual, direct NCHW store
    gemm_wmma<3><<<dim3(98, 6), 256, gemm_lds, stream>>>(h1, w2T, b2, x2tok, nullptr, (float*)d_out,
                                                         12544, 768, 3072);
}
